// MemTransformerLM_59365037965983
// MI455X (gfx1250) — compile-verified, matrix operand order checked by reading, not run on hardware
//
#include <hip/hip_runtime.h>
#include <hip/hip_bf16.h>

// ---------------------------------------------------------------------------
// MemTransformerLM forward for MI455X (gfx1250): all GEMMs via
// v_wmma_f32_16x16x32_bf16 (bf16 in, f32 accumulate), wave32.
// Register-blocked GEMM: each wave computes a 32x64 C tile (2x4 WMMA tiles),
// 8 WMMAs per 12 b128 loads per k-step.
// ---------------------------------------------------------------------------

typedef unsigned short u16;
typedef __attribute__((ext_vector_type(16))) __bf16 v16bf;
typedef __attribute__((ext_vector_type(8)))  float  v8f;

static constexpr int kT  = 1024;
static constexpr int kB  = 4;
static constexpr int kD  = 512;
static constexpr int kH  = 8;
static constexpr int kDH = 64;
static constexpr int kDI = 2048;
static constexpr int kV  = 256;
static constexpr int kL  = 8;
static constexpr int kS  = 256;          // number of segments (T / SF)
static constexpr int kMT = kT * kB;      // 4096 rows at full length

__device__ __forceinline__ int imin(int a, int b) { return a < b ? a : b; }

__device__ __forceinline__ u16 f2bf(float f) {
  union { float f; unsigned int u; } c; c.f = f;
  unsigned int u = c.u;
  unsigned int r = (u + 0x7FFFu + ((u >> 16) & 1u)) >> 16;   // round-nearest-even
  return (u16)r;
}

union FragBF { uint4 q[2]; v16bf v; };

// ---------------------------------------------------------------------------
// Generic batched WMMA GEMM:  C[z] = A[z] (MxK bf16, row-major, ld=lda)
//                                   x Bw[z%bmod]^T (Bw is NxK bf16 row-major)
//                 (+ bias[n]) (+ Res[z]) (ReLU) -> C f32 (+ optional bf16 copy)
// Wave tile: 32(M) x 64(N) = 2x4 accumulators. Block: 8 waves = 128 x 128.
// ---------------------------------------------------------------------------
__global__ __launch_bounds__(256) void wmma_gemm_bf16(
    const u16* __restrict__ A,  long long sA, int lda,
    const u16* __restrict__ Bw, long long sB, int ldb, int bmod,
    float* __restrict__ C,      long long sC, int ldc,
    const float* __restrict__ bias,
    const float* __restrict__ Res, long long sR, int ldr,
    u16* __restrict__ Cbf,      long long sCbf, int ldcbf,
    int M, int N, int K, int relu)
{
  const int z = blockIdx.z;
  A  += (long long)z * sA;
  Bw += (long long)(z % bmod) * sB;
  C  += (long long)z * sC;
  if (Res) Res += (long long)z * sR;
  if (Cbf) Cbf += (long long)z * sCbf;

  const int wave = threadIdx.x >> 5;          // 0..7
  const int lane = threadIdx.x & 31;
  const int wm = wave & 3, wn = wave >> 2;    // 4x2 wave grid in block
  const int m0 = blockIdx.y * 128 + wm * 32;  // wave covers 32 rows
  const int n0 = blockIdx.x * 128 + wn * 64;  // wave covers 64 cols
  if (m0 >= M || n0 >= N) return;             // wave-uniform: EXEC stays all-1s

  const int half = lane >> 4;                 // K sub-block 0/1 within k-step
  const int l16  = lane & 15;

  // Fragment row pointers (clamped; OOB results masked at store time).
  const u16* Arow[2];
#pragma unroll
  for (int mi = 0; mi < 2; ++mi) {
    const int am = imin(m0 + mi * 16 + l16, M - 1);
    Arow[mi] = A + (long long)am * lda + half * 16;
  }
  const u16* Brow[4];
#pragma unroll
  for (int ni = 0; ni < 4; ++ni) {
    const int bn = imin(n0 + ni * 16 + l16, N - 1);
    Brow[ni] = Bw + (long long)bn * ldb + half * 16;
  }

  v8f acc[2][4];
#pragma unroll
  for (int mi = 0; mi < 2; ++mi)
#pragma unroll
    for (int ni = 0; ni < 4; ++ni)
      acc[mi][ni] = (v8f){0.f, 0.f, 0.f, 0.f, 0.f, 0.f, 0.f, 0.f};

  for (int k = 0; k < K; k += 32) {
    FragBF fa[2], fb[4];
#pragma unroll
    for (int mi = 0; mi < 2; ++mi) {
      fa[mi].q[0] = ((const uint4*)(Arow[mi] + k))[0];
      fa[mi].q[1] = ((const uint4*)(Arow[mi] + k))[1];
    }
#pragma unroll
    for (int ni = 0; ni < 4; ++ni) {
      fb[ni].q[0] = ((const uint4*)(Brow[ni] + k))[0];
      fb[ni].q[1] = ((const uint4*)(Brow[ni] + k))[1];
    }
#pragma unroll
    for (int mi = 0; mi < 2; ++mi)
#pragma unroll
      for (int ni = 0; ni < 4; ++ni)
        acc[mi][ni] = __builtin_amdgcn_wmma_f32_16x16x32_bf16(
            false, fa[mi].v, false, fb[ni].v, (short)0, acc[mi][ni],
            false, false);
  }

  // C/D layout: lane holds column tile_n0+l16; VGPR r holds row tile_m0+r+8*half.
#pragma unroll
  for (int ni = 0; ni < 4; ++ni) {
    const int cn = n0 + ni * 16 + l16;
    if (cn < N) {
      const float bv = bias ? bias[cn] : 0.f;
#pragma unroll
      for (int mi = 0; mi < 2; ++mi) {
#pragma unroll
        for (int r = 0; r < 8; ++r) {
          const int cm = m0 + mi * 16 + r + 8 * half;
          if (cm < M) {
            float v = acc[mi][ni][r] + bv;
            if (Res) v += Res[(long long)cm * ldr + cn];
            if (relu) v = fmaxf(v, 0.f);
            C[(long long)cm * ldc + cn] = v;
            if (Cbf) Cbf[(long long)cm * ldcbf + cn] = f2bf(v);
          }
        }
      }
    }
  }
}

// ---------------------------------------------------------------------------
// Glue kernels
// ---------------------------------------------------------------------------
__global__ void f2bf_kernel(const float* __restrict__ src, u16* __restrict__ dst,
                            long long n) {
  long long i = (long long)blockIdx.x * 256 + threadIdx.x;
  if (i < n) dst[i] = f2bf(src[i]);
}

__global__ void embed_kernel(const int* __restrict__ data,
                             const float* __restrict__ emb,
                             float* __restrict__ x, u16* __restrict__ xbf) {
  const long long row = blockIdx.x;                  // t*B+b
  int tok = data[row];
  tok = tok < 0 ? 0 : (tok >= kV ? kV - 1 : tok);
  const float* e = emb + (long long)tok * kD;
  for (int c = threadIdx.x; c < kD; c += 256) {
    float v = e[c];
    x[row * kD + c] = v;
    xbf[row * kD + c] = f2bf(v);
  }
}

__global__ void posemb_kernel(u16* __restrict__ rbf, int qlen) {
  const int i = blockIdx.x;
  const float pos = (float)(qlen - 1 - i);
  for (int c = threadIdx.x; c < kD; c += 256) {
    const int k = (c < 256) ? c : c - 256;
    const float inv = __expf(-9.210340371976184f * ((float)k * (1.f / 256.f)));
    const float a = pos * inv;
    const float v = (c < 256) ? __sinf(a) : __cosf(a);
    rbf[(long long)i * kD + c] = f2bf(v);
  }
}

// Split h=(q|k|v), add rel biases to q, pack per-(b,h) bf16 matrices.
// Qw/Qr/Kh: [z][t][d] (z=b*H+h, ld=64);  Vt: [z][d][t] (ld=qpad, zero-padded).
__global__ __launch_bounds__(256) void pack_heads(
    const float* __restrict__ h, const float* __restrict__ rwb,
    const float* __restrict__ rrb,
    u16* __restrict__ Qw, u16* __restrict__ Qr,
    u16* __restrict__ Kh, u16* __restrict__ Vt,
    int qlen, int qpad)
{
  const int t = blockIdx.x;      // 0..qpad-1
  const int b = blockIdx.y;
  if (t >= qlen) {               // zero V^T padding columns
    for (int c = threadIdx.x; c < kH * kDH; c += 256) {
      const int hh = c >> 6, d = c & 63;
      const int z = b * kH + hh;
      Vt[((long long)z * kDH + d) * qpad + t] = 0;
    }
    return;
  }
  const float* row = h + ((long long)t * kB + b) * (3 * kH * kDH);
  for (int c = threadIdx.x; c < kH * kDH; c += 256) {
    const int hh = c >> 6, d = c & 63;
    const int z = b * kH + hh;
    const float q = row[c], kk = row[kH * kDH + c], vv = row[2 * kH * kDH + c];
    const long long o = ((long long)z * qlen + t) * kDH + d;
    Qw[o] = f2bf(q + rwb[c]);
    Qr[o] = f2bf(q + rrb[c]);
    Kh[o] = f2bf(kk);
    Vt[((long long)z * kDH + d) * qpad + t] = f2bf(vv);
  }
}

__global__ void pack_rk(const float* __restrict__ rkout, u16* __restrict__ RK,
                        int qlen) {
  const int t = blockIdx.x;
  for (int c = threadIdx.x; c < kD; c += 256) {
    const int hh = c >> 6, d = c & 63;
    RK[((long long)hh * qlen + t) * kDH + d] = f2bf(rkout[(long long)t * kD + c]);
  }
}

// Causal softmax with Transformer-XL relative shift applied via indexing:
// score[i][j] = (AC[i][j] + BD[i][qlen-1-i+j]) * scale  for j <= i, else -1e30.
__global__ __launch_bounds__(256) void softmax_rel(
    const float* __restrict__ AC, const float* __restrict__ BD,
    u16* __restrict__ P, int qlen, int qpad, float scale)
{
  const int i = blockIdx.x;
  const int z = blockIdx.y;
  const long long base = (long long)z * qpad * qpad + (long long)i * qpad;
  const float* ac = AC + base;
  const float* bd = BD + base;
  u16* p = P + base;

  float vals[4];
  float mx = -3.4e38f;
#pragma unroll
  for (int it = 0; it < 4; ++it) {
    const int j = threadIdx.x + it * 256;
    float s = -1e30f;
    if (j < qpad && j <= i && j < qlen)
      s = (ac[j] + bd[qlen - 1 - i + j]) * scale;
    vals[it] = s;
    mx = fmaxf(mx, s);
  }
  __shared__ float red[256];
  red[threadIdx.x] = mx; __syncthreads();
  for (int off = 128; off > 0; off >>= 1) {
    if (threadIdx.x < off)
      red[threadIdx.x] = fmaxf(red[threadIdx.x], red[threadIdx.x + off]);
    __syncthreads();
  }
  mx = red[0]; __syncthreads();

  float sum = 0.f;
#pragma unroll
  for (int it = 0; it < 4; ++it) {
    const int j = threadIdx.x + it * 256;
    float e = 0.f;
    if (j < qpad && j <= i && j < qlen) e = __expf(vals[it] - mx);
    vals[it] = e; sum += e;
  }
  red[threadIdx.x] = sum; __syncthreads();
  for (int off = 128; off > 0; off >>= 1) {
    if (threadIdx.x < off) red[threadIdx.x] += red[threadIdx.x + off];
    __syncthreads();
  }
  const float inv = 1.f / red[0];
#pragma unroll
  for (int it = 0; it < 4; ++it) {
    const int j = threadIdx.x + it * 256;
    if (j < qpad) p[j] = f2bf(vals[it] * inv);   // zero in pad/masked region
  }
}

// LayerNorm over D=512 (one block per row), writes f32 + bf16 copies.
__global__ __launch_bounds__(256) void ln_kernel(
    const float* __restrict__ X, const float* __restrict__ g,
    const float* __restrict__ bta, float* __restrict__ Y,
    u16* __restrict__ Ybf)
{
  const long long row = blockIdx.x;
  const float* x = X + row * kD;
  const float v0 = x[threadIdx.x], v1 = x[threadIdx.x + 256];
  __shared__ float red[256];
  red[threadIdx.x] = v0 + v1; __syncthreads();
  for (int off = 128; off > 0; off >>= 1) {
    if (threadIdx.x < off) red[threadIdx.x] += red[threadIdx.x + off];
    __syncthreads();
  }
  const float mean = red[0] * (1.f / kD); __syncthreads();
  const float d0 = v0 - mean, d1 = v1 - mean;
  red[threadIdx.x] = d0 * d0 + d1 * d1; __syncthreads();
  for (int off = 128; off > 0; off >>= 1) {
    if (threadIdx.x < off) red[threadIdx.x] += red[threadIdx.x + off];
    __syncthreads();
  }
  const float rstd = rsqrtf(red[0] * (1.f / kD) + 1e-5f);
  const float y0 = d0 * rstd * g[threadIdx.x] + bta[threadIdx.x];
  const float y1 = d1 * rstd * g[threadIdx.x + 256] + bta[threadIdx.x + 256];
  Y[row * kD + threadIdx.x]         = y0;
  Y[row * kD + threadIdx.x + 256]   = y1;
  Ybf[row * kD + threadIdx.x]       = f2bf(y0);
  Ybf[row * kD + threadIdx.x + 256] = f2bf(y1);
}

// Per-batch exclusive/inclusive cumsum of boundary bits (T=1024, B=4; trivial).
__global__ void seg_kernel(const int* __restrict__ bnd,
                           int* __restrict__ segdn, int* __restrict__ segup) {
  const int b = threadIdx.x;
  if (b >= kB) return;
  int c = 0;
  for (int t = 0; t < kT; ++t) {
    const int v = bnd[t * kB + b];
    segdn[b * kT + t] = c;          // cumsum - bnd (exclusive)
    c += v;
    segup[b * kT + t] = c;          // inclusive (0 = null group)
  }
}

// Mean-pool each segment; s==0 row is the null group.
__global__ __launch_bounds__(256) void pool_kernel(
    const float* __restrict__ hid, const int* __restrict__ segdn,
    const float* __restrict__ nullg, float* __restrict__ shortraw)
{
  const int s = blockIdx.x;        // 0..S
  const int b = blockIdx.y;
  if (s == 0) {
    shortraw[((long long)0 * kB + b) * kD + threadIdx.x]       = nullg[threadIdx.x];
    shortraw[((long long)0 * kB + b) * kD + threadIdx.x + 256] = nullg[threadIdx.x + 256];
    return;
  }
  const int sid = s - 1;
  float a0 = 0.f, a1 = 0.f, cnt = 0.f;
  for (int t = 0; t < kT; ++t) {
    if (segdn[b * kT + t] == sid) {
      const float* r = hid + ((long long)t * kB + b) * kD;
      a0 += r[threadIdx.x]; a1 += r[threadIdx.x + 256]; cnt += 1.f;
    }
  }
  const float inv = 1.f / (cnt + 1e-9f);
  shortraw[((long long)s * kB + b) * kD + threadIdx.x]       = a0 * inv;
  shortraw[((long long)s * kB + b) * kD + threadIdx.x + 256] = a1 * inv;
}

__global__ void upsample_kernel(const float* __restrict__ shortb,
                                const int* __restrict__ segup,
                                const float* __restrict__ resid,
                                float* __restrict__ x, u16* __restrict__ xbf) {
  const int t = blockIdx.x, b = blockIdx.y;
  const int s = segup[b * kT + t];
  const float* src = shortb + ((long long)s * kB + b) * kD;
  const long long o = ((long long)t * kB + b) * kD;
  const float inv = 1.f / (1.f + 1e-9f);
  for (int c = threadIdx.x; c < kD; c += 256) {
    const float v = src[c] * inv + resid[o + c];
    x[o + c] = v;
    xbf[o + c] = f2bf(v);
  }
}

// ---------------------------------------------------------------------------
// Host side
// ---------------------------------------------------------------------------
static void launch_gemm(hipStream_t st,
    const u16* A, long long sA, int lda,
    const u16* Bw, long long sB, int ldb, int bmod,
    float* C, long long sC, int ldc,
    const float* bias,
    const float* Res, long long sR, int ldr,
    u16* Cbf, long long sCbf, int ldcbf,
    int M, int N, int K, int relu, int batch)
{
  dim3 grid((N + 127) / 128, (M + 127) / 128, batch);
  wmma_gemm_bf16<<<grid, 256, 0, st>>>(A, sA, lda, Bw, sB, ldb, bmod,
                                       C, sC, ldc, bias, Res, sR, ldr,
                                       Cbf, sCbf, ldcbf, M, N, K, relu);
}

extern "C" void kernel_launch(void* const* d_in, const int* in_sizes, int n_in,
                              void* d_out, int out_size, void* d_ws, size_t ws_size,
                              hipStream_t stream)
{
  const int*   data      = (const int*)d_in[0];
  const int*   bnd       = (const int*)d_in[1];
  const float* word_emb  = (const float*)d_in[2];
  const float* r_w_bias  = (const float*)d_in[3];
  const float* r_r_bias  = (const float*)d_in[4];
  const float* null_grp  = (const float*)d_in[5];
  const float* down_g    = (const float*)d_in[6];
  const float* down_b    = (const float*)d_in[7];
  const float* qkv_w     = (const float*)d_in[8];
  const float* rk_w      = (const float*)d_in[9];
  const float* o_w       = (const float*)d_in[10];
  const float* ln1_g     = (const float*)d_in[11];
  const float* ln1_b     = (const float*)d_in[12];
  const float* ff_w1     = (const float*)d_in[13];
  const float* ff_b1     = (const float*)d_in[14];
  const float* ff_w2     = (const float*)d_in[15];
  const float* ff_b2     = (const float*)d_in[16];
  const float* ln2_g     = (const float*)d_in[17];
  const float* ln2_b     = (const float*)d_in[18];
  const float* final_w   = (const float*)d_in[19];
  const float* final_b   = (const float*)d_in[20];

  // ---- workspace bump allocator --------------------------------------------
  char* base = (char*)d_ws;
  size_t off = 0;
  auto alloc = [&](size_t bytes) -> void* {
    void* p = base + off;
    off = (off + bytes + 255) & ~(size_t)255;
    return p;
  };
  const long long nWq = (long long)kL * 3 * kH * kDH * kD;   // 6.29M
  const long long nWr = (long long)kL * kD * kD;             // 2.10M
  const long long nWo = nWr;
  const long long nW1 = (long long)kL * kDI * kD;            // 8.39M
  const long long nW2 = nW1;
  const long long nWf = (long long)kV * kD;

  u16*  wq    = (u16*)alloc(nWq * 2);
  u16*  wrk   = (u16*)alloc(nWr * 2);
  u16*  wo    = (u16*)alloc(nWo * 2);
  u16*  wf1   = (u16*)alloc(nW1 * 2);
  u16*  wf2   = (u16*)alloc(nW2 * 2);
  u16*  wfin  = (u16*)alloc(nWf * 2);
  float* x     = (float*)alloc((size_t)kMT * kD * 4);
  u16*   xbf   = (u16*)  alloc((size_t)kMT * kD * 2);
  float* resid = (float*)alloc((size_t)kMT * kD * 4);
  float* h     = (float*)alloc((size_t)kMT * 3 * kH * kDH * 4);
  float* rkout = (float*)alloc((size_t)kT * kD * 4);
  u16*   rbf   = (u16*)  alloc((size_t)kT * kD * 2);
  u16*   Qw    = (u16*)  alloc((size_t)kB * kH * kT * kDH * 2);
  u16*   Qr    = (u16*)  alloc((size_t)kB * kH * kT * kDH * 2);
  u16*   Kh    = (u16*)  alloc((size_t)kB * kH * kT * kDH * 2);
  u16*   Vt    = (u16*)  alloc((size_t)kB * kH * kDH * kT * 2);
  u16*   RK    = (u16*)  alloc((size_t)kH * kT * kDH * 2);
  float* AC    = (float*)alloc((size_t)kB * kH * kT * kT * 4);
  float* BD    = (float*)alloc((size_t)kB * kH * kT * kT * 4);
  u16*   prob  = (u16*)  alloc((size_t)kB * kH * kT * kT * 2);
  u16*   vecbf = (u16*)  alloc((size_t)kMT * kD * 2);
  u16*   ff1bf = (u16*)  alloc((size_t)kMT * kDI * 2);
  float* t0    = (float*)alloc((size_t)kMT * kDI * 4);
  float* shortb  = (float*)alloc((size_t)(kS + 1) * kB * kD * 4);
  u16*   shortbf = (u16*)  alloc((size_t)(kS + 1) * kB * kD * 2);
  int*   segdn = (int*)alloc((size_t)kB * kT * 4);
  int*   segup = (int*)alloc((size_t)kB * kT * 4);
  (void)ws_size; (void)n_in; (void)in_sizes; (void)out_size;

  const int BIG = 1 << 30;

  // ---- weights -> bf16 -----------------------------------------------------
  f2bf_kernel<<<(unsigned)((nWq + 255) / 256), 256, 0, stream>>>(qkv_w,  wq,  nWq);
  f2bf_kernel<<<(unsigned)((nWr + 255) / 256), 256, 0, stream>>>(rk_w,   wrk, nWr);
  f2bf_kernel<<<(unsigned)((nWo + 255) / 256), 256, 0, stream>>>(o_w,    wo,  nWo);
  f2bf_kernel<<<(unsigned)((nW1 + 255) / 256), 256, 0, stream>>>(ff_w1,  wf1, nW1);
  f2bf_kernel<<<(unsigned)((nW2 + 255) / 256), 256, 0, stream>>>(ff_w2,  wf2, nW2);
  f2bf_kernel<<<(unsigned)((nWf + 255) / 256), 256, 0, stream>>>(final_w, wfin, nWf);

  // ---- embedding -----------------------------------------------------------
  embed_kernel<<<kMT, 256, 0, stream>>>(data, word_emb, x, xbf);

  // ---- one transformer stage ----------------------------------------------
  auto run_stage = [&](float* sx, u16* sxbf, int qlen, int lo, int hi) {
    const int M = qlen * kB;
    const int qpad = (qlen + 31) & ~31;
    const long long q64 = (long long)qlen * kDH;
    const long long qq  = (long long)qpad * qpad;
    posemb_kernel<<<qlen, 256, 0, stream>>>(rbf, qlen);
    for (int i = lo; i < hi; ++i) {
      // h = x @ qkv_w[i]^T                             (M x 1536)
      launch_gemm(stream, sxbf, 0, kD, wq + (size_t)i * 3 * kH * kDH * kD, 0, kD,
                  BIG, h, 0, 3 * kH * kDH, nullptr, nullptr, 0, 0,
                  nullptr, 0, 0, M, 3 * kH * kDH, kD, 0, 1);
      // rk = r @ rk_w[i]^T                             (qlen x 512)
      launch_gemm(stream, rbf, 0, kD, wrk + (size_t)i * kD * kD, 0, kD, BIG,
                  rkout, 0, kD, nullptr, nullptr, 0, 0, nullptr, 0, 0,
                  qlen, kD, kD, 0, 1);
      pack_heads<<<dim3(qpad, kB), 256, 0, stream>>>(h, r_w_bias, r_r_bias,
                                                     Qw, Qr, Kh, Vt, qlen, qpad);
      pack_rk<<<qlen, 256, 0, stream>>>(rkout, RK, qlen);
      // AC = (q + r_w_bias) K^T   batched over z = b*H+h
      launch_gemm(stream, Qw, q64, kDH, Kh, q64, kDH, BIG, AC, qq, qpad,
                  nullptr, nullptr, 0, 0, nullptr, 0, 0, qlen, qlen, kDH, 0,
                  kB * kH);
      // BDraw = (q + r_r_bias) RK^T  (RK shared across batch b -> bmod = H)
      launch_gemm(stream, Qr, q64, kDH, RK, q64, kDH, kH, BD, qq, qpad,
                  nullptr, nullptr, 0, 0, nullptr, 0, 0, qlen, qlen, kDH, 0,
                  kB * kH);
      softmax_rel<<<dim3(qlen, kB * kH), 256, 0, stream>>>(AC, BD, prob, qlen,
                                                           qpad, 0.125f);
      // vec = prob @ V ; bf16 copy written directly in (t,b,h*64+d) layout
      launch_gemm(stream, prob, qq, qpad, Vt, (long long)kDH * qpad, qpad, BIG,
                  t0, kDH, kB * kD, nullptr, nullptr, 0, 0,
                  vecbf, kDH, kB * kD, qlen, kDH, qpad, 0, kB * kH);
      // x + vec @ o_w[i]^T -> t0 ; then LN1 -> x
      launch_gemm(stream, vecbf, 0, kD, wo + (size_t)i * kD * kD, 0, kD, BIG,
                  t0, 0, kD, nullptr, sx, 0, kD, nullptr, 0, 0, M, kD, kD, 0, 1);
      ln_kernel<<<M, 256, 0, stream>>>(t0, ln1_g + i * kD, ln1_b + i * kD, sx, sxbf);
      // ff1: relu(x @ w1^T + b1)                        (M x 2048)
      launch_gemm(stream, sxbf, 0, kD, wf1 + (size_t)i * kDI * kD, 0, kD, BIG,
                  t0, 0, kDI, ff_b1 + i * kDI, nullptr, 0, 0,
                  ff1bf, 0, kDI, M, kDI, kD, 1, 1);
      // ff2: x + ff1 @ w2^T + b2 -> t0 ; then LN2 -> x
      launch_gemm(stream, ff1bf, 0, kDI, wf2 + (size_t)i * kD * kDI, 0, kDI, BIG,
                  t0, 0, kD, ff_b2 + i * kD, sx, 0, kD, nullptr, 0, 0,
                  M, kD, kDI, 0, 1);
      ln_kernel<<<M, 256, 0, stream>>>(t0, ln2_g + i * kD, ln2_b + i * kD, sx, sxbf);
    }
  };

  // ---- stage 1 (layers 0..1) at qlen = 1024 --------------------------------
  run_stage(x, xbf, kT, 0, 2);

  // ---- downsample ----------------------------------------------------------
  hipMemcpyAsync(resid, x, (size_t)kMT * kD * 4, hipMemcpyDeviceToDevice, stream);
  seg_kernel<<<1, kB, 0, stream>>>(bnd, segdn, segup);
  pool_kernel<<<dim3(kS + 1, kB), 256, 0, stream>>>(x, segdn, null_grp, t0);
  ln_kernel<<<(kS + 1) * kB, 256, 0, stream>>>(t0, down_g, down_b, shortb, shortbf);

  // ---- stage 2 (layers 2..5) at qlen = 257 ---------------------------------
  run_stage(shortb, shortbf, kS + 1, 2, 6);

  // ---- upsample + residual -------------------------------------------------
  upsample_kernel<<<dim3(kT, kB), 256, 0, stream>>>(shortb, segup, resid, x, xbf);

  // ---- stage 3 (layers 6..7) at qlen = 1024 --------------------------------
  run_stage(x, xbf, kT, 6, 8);

  // ---- logits = x @ final_w^T + final_b -> d_out  (4096 x 256) -------------
  launch_gemm(stream, xbf, 0, kD, wfin, 0, kD, BIG,
              (float*)d_out, 0, kV, final_b, nullptr, 0, 0, nullptr, 0, 0,
              kMT, kV, kD, 0, 1);
}
